// LSTMAttention_11699490914803
// MI455X (gfx1250) — matrix operand order repaired and checked by reading, not compiled, not run on hardware
//
#include <hip/hip_runtime.h>

// ---------------------------------------------------------------------------
// LSTM + attention, B=256, SEQ=128, IN=512, H=512, on MI455X (gfx1250).
// bf16 WMMA (v_wmma_f32_16x16x32_bf16) for the big gate GEMM and the
// attention-logit GEMM; f32 VALU for the batched attn@x einsum (L2-resident x).
// ---------------------------------------------------------------------------

typedef __attribute__((ext_vector_type(16))) __bf16 v16bf;
typedef __attribute__((ext_vector_type(8)))  __bf16 v8bf;
typedef __attribute__((ext_vector_type(8)))  float  v8f;

union V8F { v8f v; float f[8]; };

#define Bsz  256
#define SEQ  128
#define IN   512
#define Hdim 512
#define KCAT 1024   // IN + H
#define N4H  2048   // 4*H

__device__ __forceinline__ float sigm(float x) {
    return 1.0f / (1.0f + __expf(-x));
}

__device__ __forceinline__ v16bf cat16(v8bf lo, v8bf hi) {
    return __builtin_shufflevector(lo, hi, 0,1,2,3,4,5,6,7,8,9,10,11,12,13,14,15);
}

// ---------------------------------------------------------------------------
// Kernel 1: one-time repack.
//  - Wt_swz: B-fragments of Wcat^T (K=1024, N=2048), fragment-major:
//      frag f = kt*128 + nt ; lane holds col n = nt*16 + (lane&15),
//      K = kt*32 + (lane>>4)*16 + e  (e = 0..15), contiguous 32B per lane.
//  - Wa_swz: same for Wa^T (K=512, N=128): frag = kt*8 + nt.
//  - bias = b_ih + b_hh.
// ---------------------------------------------------------------------------
__global__ __launch_bounds__(256)
void k_repack(const float* __restrict__ W_ih, const float* __restrict__ W_hh,
              const float* __restrict__ Wa,
              const float* __restrict__ b_ih, const float* __restrict__ b_hh,
              __bf16* __restrict__ Wt_swz, __bf16* __restrict__ Wa_swz,
              float* __restrict__ bias)
{
    const int idx = blockIdx.x * 256 + threadIdx.x;
    const int WT_T = 32 * 128 * 32;   // 131072 lane-slots
    const int WA_T = 16 * 8 * 32;     // 4096 lane-slots

    if (idx < WT_T) {
        const int lane = idx & 31;
        const int frag = idx >> 5;
        const int nt   = frag & 127;
        const int kt   = frag >> 7;
        const int n    = nt * 16 + (lane & 15);
        const int kb   = kt * 32 + (lane >> 4) * 16;
        __bf16* dst = Wt_swz + (size_t)idx * 16;
        #pragma unroll
        for (int e = 0; e < 16; ++e) {
            const int k = kb + e;
            const float v = (k < IN) ? W_ih[(size_t)n * IN + k]
                                     : W_hh[(size_t)n * Hdim + (k - IN)];
            dst[e] = (__bf16)v;
        }
    } else if (idx < WT_T + WA_T) {
        const int j    = idx - WT_T;
        const int lane = j & 31;
        const int frag = j >> 5;
        const int nt   = frag & 7;
        const int kt   = frag >> 3;
        const int n    = nt * 16 + (lane & 15);
        const int kb   = kt * 32 + (lane >> 4) * 16;
        __bf16* dst = Wa_swz + (size_t)j * 16;
        #pragma unroll
        for (int e = 0; e < 16; ++e)
            dst[e] = (__bf16)Wa[(size_t)n * Hdim + kb + e];
    } else if (idx < WT_T + WA_T + N4H) {
        const int d = idx - WT_T - WA_T;
        bias[d] = b_ih[d] + b_hh[d];
    }
}

// ---------------------------------------------------------------------------
// Kernel 2: initial cell on zero x/h/c  ->  gates == bias (batch-invariant).
//   c0 = sigm(i)*tanh(g);  h0 = sigm(o)*tanh(c0)   (f*c term is f*0 = 0)
// ---------------------------------------------------------------------------
__global__ __launch_bounds__(512)
void k_init_state(const float* __restrict__ bias,
                  float* __restrict__ c_f32, __bf16* __restrict__ c_bf,
                  __bf16* __restrict__ h0_bf)
{
    const int d = threadIdx.x;     // 0..511
    const int b = blockIdx.x;      // 0..255
    const float iv = sigm(bias[d]);
    const float gv = tanhf(bias[2 * Hdim + d]);
    const float ov = sigm(bias[3 * Hdim + d]);
    const float c0 = iv * gv;
    const float h0 = ov * tanhf(c0);
    c_f32[(size_t)b * Hdim + d] = c0;
    c_bf [(size_t)b * Hdim + d] = (__bf16)c0;
    h0_bf[(size_t)b * Hdim + d] = (__bf16)h0;
}

// ---------------------------------------------------------------------------
// Kernel 3: per-step attention + xi.
//  grid = (16 m-tiles, 4 d-quarters), block = 256 (8 waves).
//  Phase 1: logits[16x128] = c_bf(m-tile) @ Wa^T + ba via WMMA (wave w -> cols
//           16w..16w+15), result staged to LDS.
//  Phase 2: row softmax in LDS (threads 0..15, serial over 128 — tiny).
//  Phase 3: xi[row, d0..d0+7] = sum_s attn[row,s] * x[row,s,d]  (f32 FMA,
//           x streamed from L2), stored bf16 into Xi for the gate GEMM.
// ---------------------------------------------------------------------------
__global__ __launch_bounds__(256)
void k_attn_xi(const __bf16* __restrict__ c_bf, const __bf16* __restrict__ Wa_swz,
               const float* __restrict__ ba, const float* __restrict__ x,
               __bf16* __restrict__ Xi)
{
    __shared__ float att[16][SEQ + 1];

    const int m0   = blockIdx.x * 16;
    const int w    = threadIdx.x >> 5;
    const int lane = threadIdx.x & 31;
    const int hf   = lane >> 4;       // which lane-half
    const int lcol = lane & 15;

    // ---- Phase 1: WMMA logits ------------------------------------------------
    V8F acc;
    {
        const float bav = ba[w * 16 + lcol];
        #pragma unroll
        for (int r = 0; r < 8; ++r) acc.f[r] = bav;
    }
    const __bf16* arow = c_bf + (size_t)(m0 + lcol) * Hdim + hf * 16;
    for (int kt = 0; kt < Hdim / 32; ++kt) {
        v8bf lo = *(const v8bf*)(arow + kt * 32);
        v8bf hi = *(const v8bf*)(arow + kt * 32 + 8);
        v16bf a = cat16(lo, hi);
        v16bf b = *(const v16bf*)(Wa_swz + ((size_t)(kt * 8 + w) * 32 + lane) * 16);
        acc.v = __builtin_amdgcn_wmma_f32_16x16x32_bf16(
                    false, a, false, b, (short)0, acc.v, false, false);
    }
    #pragma unroll
    for (int r = 0; r < 8; ++r)
        att[hf ? 8 + r : r][w * 16 + lcol] = acc.f[r];
    __syncthreads();

    // ---- Phase 2: softmax ----------------------------------------------------
    if (threadIdx.x < 16) {
        const int row = threadIdx.x;
        float mx = -3.402823466e+38f;
        for (int s = 0; s < SEQ; ++s) mx = fmaxf(mx, att[row][s]);
        float sum = 0.0f;
        for (int s = 0; s < SEQ; ++s) {
            const float e = __expf(att[row][s] - mx);
            att[row][s] = e;
            sum += e;
        }
        const float inv = 1.0f / sum;
        for (int s = 0; s < SEQ; ++s) att[row][s] *= inv;
    }
    __syncthreads();

    // ---- Phase 3: xi = attn @ x[batch] --------------------------------------
    const int row = threadIdx.x >> 4;                 // 0..15  (batch row in tile)
    const int d0  = blockIdx.y * 128 + (threadIdx.x & 15) * 8;
    float4 a0 = {0.f, 0.f, 0.f, 0.f}, a1 = {0.f, 0.f, 0.f, 0.f};
    const float* xr = x + ((size_t)(m0 + row) * SEQ) * IN + d0;
    for (int s = 0; s < SEQ; ++s) {
        const float aw = att[row][s];
        const float4* xp = (const float4*)(xr + (size_t)s * IN);
        if (s + 2 < SEQ) __builtin_prefetch(xr + (size_t)(s + 2) * IN, 0, 0);
        const float4 v0 = xp[0], v1 = xp[1];
        a0.x += aw * v0.x; a0.y += aw * v0.y; a0.z += aw * v0.z; a0.w += aw * v0.w;
        a1.x += aw * v1.x; a1.y += aw * v1.y; a1.z += aw * v1.z; a1.w += aw * v1.w;
    }
    __bf16 tmp[8];
    tmp[0] = (__bf16)a0.x; tmp[1] = (__bf16)a0.y; tmp[2] = (__bf16)a0.z; tmp[3] = (__bf16)a0.w;
    tmp[4] = (__bf16)a1.x; tmp[5] = (__bf16)a1.y; tmp[6] = (__bf16)a1.z; tmp[7] = (__bf16)a1.w;
    *(v8bf*)(Xi + (size_t)(m0 + row) * Hdim + d0) = *(v8bf*)tmp;
}

// ---------------------------------------------------------------------------
// Kernel 4: fused gate GEMM + LSTM cell.
//  gates[256,2048] = [Xi | h_in] @ Wcat^T + bias; each wave owns rows
//  m0..m0+15 and ONE 16-col slice of all four gate groups (n0, n0+512,
//  n0+1024, n0+1536), so i/f/g/o line up per accumulator lane and the cell
//  update is fully in-register. Writes c (f32+bf16), h_out (bf16, ping-pong)
//  and the step's output row of hs.
//  grid = 64 (8 m-blocks x 8 n-blocks), block = 256 (2x4 waves of 16x16).
// ---------------------------------------------------------------------------
__global__ __launch_bounds__(256)
void k_gemm_cell(const __bf16* __restrict__ Xi, const __bf16* __restrict__ Hin,
                 const __bf16* __restrict__ Wt_swz, const float* __restrict__ bias,
                 float* __restrict__ c_f32, __bf16* __restrict__ c_bf,
                 __bf16* __restrict__ Hout, float* __restrict__ out, int t)
{
    const int mBlk = blockIdx.x & 7;
    const int nBlk = blockIdx.x >> 3;
    const int w    = threadIdx.x >> 5;
    const int lane = threadIdx.x & 31;
    const int hf   = lane >> 4;
    const int lcol = lane & 15;
    const int m0   = mBlk * 32 + (w & 1) * 16;
    const int n0   = nBlk * 64 + (w >> 1) * 16;     // in [0, 512)
    const int col  = n0 + lcol;

    V8F aI, aF, aG, aO;
    {
        const float bI = bias[col];
        const float bF = bias[Hdim + col];
        const float bG = bias[2 * Hdim + col];
        const float bO = bias[3 * Hdim + col];
        #pragma unroll
        for (int r = 0; r < 8; ++r) { aI.f[r] = bI; aF.f[r] = bF; aG.f[r] = bG; aO.f[r] = bO; }
    }

    const __bf16* xrow = Xi  + (size_t)(m0 + lcol) * Hdim + hf * 16;
    const __bf16* hrow = Hin + (size_t)(m0 + lcol) * Hdim + hf * 16;
    const int nt = n0 >> 4;
    // per-gate fragment bases; fragment stride in K-tiles is 128*32*16 elems
    const __bf16* bI16 = Wt_swz + (((size_t)(0 * 32 + nt)) * 32 + lane) * 16;
    const __bf16* bF16 = Wt_swz + (((size_t)(1 * 32 + nt)) * 32 + lane) * 16;
    const __bf16* bG16 = Wt_swz + (((size_t)(2 * 32 + nt)) * 32 + lane) * 16;
    const __bf16* bO16 = Wt_swz + (((size_t)(3 * 32 + nt)) * 32 + lane) * 16;
    const size_t kstride = (size_t)128 * 32 * 16;

    for (int kt = 0; kt < KCAT / 32; ++kt) {
        const __bf16* src = (kt < 16) ? xrow : hrow;
        const int kl = (kt & 15) * 32;
        v8bf lo = *(const v8bf*)(src + kl);
        v8bf hi = *(const v8bf*)(src + kl + 8);
        v16bf a = cat16(lo, hi);
        const size_t ko = (size_t)kt * kstride;
        v16bf fI = *(const v16bf*)(bI16 + ko);
        v16bf fF = *(const v16bf*)(bF16 + ko);
        v16bf fG = *(const v16bf*)(bG16 + ko);
        v16bf fO = *(const v16bf*)(bO16 + ko);
        aI.v = __builtin_amdgcn_wmma_f32_16x16x32_bf16(false, a, false, fI, (short)0, aI.v, false, false);
        aF.v = __builtin_amdgcn_wmma_f32_16x16x32_bf16(false, a, false, fF, (short)0, aF.v, false, false);
        aG.v = __builtin_amdgcn_wmma_f32_16x16x32_bf16(false, a, false, fG, (short)0, aG.v, false, false);
        aO.v = __builtin_amdgcn_wmma_f32_16x16x32_bf16(false, a, false, fO, (short)0, aO.v, false, false);
    }

    // ---- in-register LSTM cell ----------------------------------------------
    #pragma unroll
    for (int r = 0; r < 8; ++r) {
        const int m = m0 + (hf ? 8 + r : r);
        const size_t ci = (size_t)m * Hdim + col;
        const float cprev = c_f32[ci];
        const float iv = sigm(aI.f[r]);
        const float fv = sigm(aF.f[r]);
        const float gv = tanhf(aG.f[r]);
        const float ov = sigm(aO.f[r]);
        const float cn = fv * cprev + iv * gv;
        const float hn = ov * tanhf(cn);
        c_f32[ci] = cn;
        c_bf [ci] = (__bf16)cn;
        Hout [ci] = (__bf16)hn;
        out[((size_t)m * SEQ + t) * Hdim + col] = hn;
    }
}

// ---------------------------------------------------------------------------
extern "C" void kernel_launch(void* const* d_in, const int* in_sizes, int n_in,
                              void* d_out, int out_size, void* d_ws, size_t ws_size,
                              hipStream_t stream)
{
    const float* x    = (const float*)d_in[0];   // [256,128,512]
    const float* W_ih = (const float*)d_in[1];   // [2048,512]
    const float* W_hh = (const float*)d_in[2];   // [2048,512]
    const float* b_ih = (const float*)d_in[3];   // [2048]
    const float* b_hh = (const float*)d_in[4];   // [2048]
    const float* Wa   = (const float*)d_in[5];   // [128,512]
    const float* ba   = (const float*)d_in[6];   // [128]
    float* out = (float*)d_out;                  // [256,128,512]

    char* ws = (char*)d_ws;
    size_t off = 0;
    __bf16* Wt_swz = (__bf16*)(ws + off); off += (size_t)KCAT * N4H * 2;   // 4 MB
    __bf16* Wa_swz = (__bf16*)(ws + off); off += (size_t)Hdim * SEQ * 2;   // 128 KB
    float*  bias   = (float*) (ws + off); off += (size_t)N4H * 4;          // 8 KB
    __bf16* Xi     = (__bf16*)(ws + off); off += (size_t)Bsz * Hdim * 2;   // 256 KB
    __bf16* h0     = (__bf16*)(ws + off); off += (size_t)Bsz * Hdim * 2;   // 256 KB
    __bf16* h1     = (__bf16*)(ws + off); off += (size_t)Bsz * Hdim * 2;   // 256 KB
    float*  c_f32  = (float*) (ws + off); off += (size_t)Bsz * Hdim * 4;   // 512 KB
    __bf16* c_bf   = (__bf16*)(ws + off); off += (size_t)Bsz * Hdim * 2;   // 256 KB
    (void)ws_size; (void)in_sizes; (void)n_in; (void)out_size;

    k_repack<<<536, 256, 0, stream>>>(W_ih, W_hh, Wa, b_ih, b_hh,
                                      Wt_swz, Wa_swz, bias);
    k_init_state<<<Bsz, Hdim, 0, stream>>>(bias, c_f32, c_bf, h0);

    for (int t = 0; t < SEQ; ++t) {
        k_attn_xi<<<dim3(16, 4), 256, 0, stream>>>(c_bf, Wa_swz, ba, x, Xi);
        const __bf16* hin  = (t & 1) ? h1 : h0;
        __bf16*       hout = (t & 1) ? h0 : h1;
        k_gemm_cell<<<64, 256, 0, stream>>>(Xi, hin, Wt_swz, bias,
                                            c_f32, c_bf, hout, out, t);
    }
}
